// RiemannianGNN_41558103556268
// MI455X (gfx1250) — compile-verified, matrix operand order checked by reading, not run on hardware
//
#include <hip/hip_runtime.h>

#define NN   50000
#define DEG  16
#define DD   128
#define EPSF 1e-5f

typedef __attribute__((ext_vector_type(16))) _Float16 v16h;
typedef __attribute__((ext_vector_type(8)))  float    v8f;
typedef __attribute__((ext_vector_type(4)))  float    f32x4;
typedef __attribute__((ext_vector_type(8)))  _Float16 h8;
typedef __attribute__((ext_vector_type(4)))  _Float16 h4;

// ---------------------------------------------------------------------------
// prep: y = (do_log ? logmap0(x) : x) * mask        one wave32 per node row
// ---------------------------------------------------------------------------
__global__ __launch_bounds__(256) void prep_kernel(const float* __restrict__ xin,
                                                   const float* __restrict__ mask,
                                                   float* __restrict__ y, int do_log) {
  int node = (int)((blockIdx.x * blockDim.x + threadIdx.x) >> 5);
  int lane = threadIdx.x & 31;
  if (node >= NN) return;
  const float* row = xin + (size_t)node * DD;
  f32x4 v = *(const f32x4*)(row + lane * 4);
  float s = v.x * v.x + v.y * v.y + v.z * v.z + v.w * v.w;
  #pragma unroll
  for (int m = 16; m; m >>= 1) s += __shfl_xor(s, m, 32);
  float scale = mask[node];
  if (do_log) {
    float n  = sqrtf(s);
    float nc = fminf(fmaxf(n, EPSF), 1.0f - EPSF);
    scale *= atanhf(nc) / nc;
  }
  v *= scale;
  *(f32x4*)(y + (size_t)node * DD + lane * 4) = v;
}

// ---------------------------------------------------------------------------
// wcvt: WT[n*128+k] = (f16) W[layer][k*128+n]   (column-major B for WMMA)
// blockIdx.x: 0 = pos relation, 1 = neg relation
// ---------------------------------------------------------------------------
__global__ __launch_bounds__(256) void wcvt_kernel(const float* __restrict__ Wpos,
                                                   const float* __restrict__ Wneg,
                                                   _Float16* __restrict__ WTpos,
                                                   _Float16* __restrict__ WTneg, int layer) {
  const float*  W  = (blockIdx.x ? Wneg : Wpos) + (size_t)layer * DD * DD;
  _Float16*     WT = blockIdx.x ? WTneg : WTpos;
  for (int i = threadIdx.x; i < DD * DD; i += blockDim.x) {
    int k = i >> 7, n = i & 127;
    WT[(size_t)n * DD + k] = (_Float16)W[i];
  }
}

// ---------------------------------------------------------------------------
// gemm: msgP = (f16)(X @ Wpos), msgN = (f16)(X @ Wneg)
// One wave per 16-row M-tile; 8 N-tiles x 4 K-steps of v_wmma_f32_16x16x32_f16
// per relation (64 WMMAs / wave). X is f32, converted to A-fragments in-reg.
// ---------------------------------------------------------------------------
__global__ __launch_bounds__(256) void gemm_kernel(const float* __restrict__ X,
                                                   const _Float16* __restrict__ WTp,
                                                   const _Float16* __restrict__ WTn,
                                                   _Float16* __restrict__ msgP,
                                                   _Float16* __restrict__ msgN) {
  const int lane  = threadIdx.x & 31;
  const int wave  = threadIdx.x >> 5;
  const int tile  = blockIdx.x * 8 + wave;
  if (tile >= NN / 16) return;                 // uniform per wave: EXEC stays all-ones
  const int m0    = tile * 16;
  const int half_ = lane >> 4;                 // lane group 0 / 1
  const int l16   = lane & 15;
  const float* arow = X + (size_t)(m0 + l16) * DD;

  #pragma unroll
  for (int mat = 0; mat < 2; ++mat) {
    const _Float16* WT  = mat ? WTn : WTp;
    _Float16*       msg = mat ? msgN : msgP;
    v8f acc[8] = {};
    #pragma unroll
    for (int k0 = 0; k0 < DD; k0 += 32) {
      // 16-bit A fragment layout: lanes 0-15 K={k0..k0+7, k0+16..k0+23},
      // lanes 16-31 K shifted by +8 (ISA 7.12.2, 16-bit A 16x32).
      const int kb = k0 + half_ * 8;
      f32x4 a0 = *(const f32x4*)(arow + kb);
      f32x4 a1 = *(const f32x4*)(arow + kb + 4);
      f32x4 a2 = *(const f32x4*)(arow + kb + 16);
      f32x4 a3 = *(const f32x4*)(arow + kb + 20);
      v16h afrag;
      #pragma unroll
      for (int i = 0; i < 4; ++i) {
        afrag[i]      = (_Float16)a0[i];
        afrag[4 + i]  = (_Float16)a1[i];
        afrag[8 + i]  = (_Float16)a2[i];
        afrag[12 + i] = (_Float16)a3[i];
      }
      #pragma unroll
      for (int nt = 0; nt < 8; ++nt) {
        // B fragment: column (nt*16 + l16), same K pattern; WT is col-major so
        // this is two contiguous 16B loads.
        const _Float16* bcol = WT + (size_t)(nt * 16 + l16) * DD + kb;
        h8 b0 = *(const h8*)(bcol);
        h8 b1 = *(const h8*)(bcol + 16);
        v16h bfrag;
        #pragma unroll
        for (int i = 0; i < 8; ++i) { bfrag[i] = b0[i]; bfrag[8 + i] = b1[i]; }
        acc[nt] = __builtin_amdgcn_wmma_f32_16x16x32_f16(
            /*neg_a=*/false, afrag, /*neg_b=*/false, bfrag,
            /*c_mod=*/(short)0, acc[nt], /*reuse_a=*/false, /*reuse_b=*/false);
      }
    }
    // C/D layout: VGPR r holds M = (half_? 8:0)+r, N = l16.
    #pragma unroll
    for (int nt = 0; nt < 8; ++nt) {
      #pragma unroll
      for (int r = 0; r < 8; ++r) {
        int m = m0 + half_ * 8 + r;
        msg[(size_t)m * DD + nt * 16 + l16] = (_Float16)acc[nt][r];
      }
    }
  }
}

// ---------------------------------------------------------------------------
// gather: combined = sum_j w+ * msgP[adj] + w- * msgN[adj];
//         out = relu(expmap0(combined * mask)) * mask      one wave per node
// ---------------------------------------------------------------------------
__global__ __launch_bounds__(256) void gather_kernel(const int* __restrict__ adj,
                                                     const float* __restrict__ weight,
                                                     const float* __restrict__ mask,
                                                     const _Float16* __restrict__ msgP,
                                                     const _Float16* __restrict__ msgN,
                                                     float* __restrict__ out) {
  int node = (int)((blockIdx.x * blockDim.x + threadIdx.x) >> 5);
  int lane = threadIdx.x & 31;
  if (node >= NN) return;
  f32x4 acc = {0.f, 0.f, 0.f, 0.f};
  #pragma unroll
  for (int j = 0; j < DEG; ++j) {
    float w = weight[node * DEG + j];
    int   a = adj[node * DEG + j];
    if (j + 1 < DEG) {  // prefetch next neighbor row (global_prefetch_b8)
      float wn = weight[node * DEG + j + 1];
      int   an = adj[node * DEG + j + 1];
      const _Float16* pr = (wn > 0.f ? msgP : msgN) + (size_t)an * DD + lane * 4;
      __builtin_prefetch(pr, 0, 3);
    }
    if (w > 0.f) {
      h4 h = *(const h4*)(msgP + (size_t)a * DD + lane * 4);
      #pragma unroll
      for (int i = 0; i < 4; ++i) acc[i] += w * (float)h[i];
    } else if (w < 0.f) {
      h4 h = *(const h4*)(msgN + (size_t)a * DD + lane * 4);
      #pragma unroll
      for (int i = 0; i < 4; ++i) acc[i] += (-w) * (float)h[i];
    }
  }
  float mk = mask[node];
  acc *= mk;
  float s = acc[0]*acc[0] + acc[1]*acc[1] + acc[2]*acc[2] + acc[3]*acc[3];
  #pragma unroll
  for (int m = 16; m; m >>= 1) s += __shfl_xor(s, m, 32);
  float n     = sqrtf(s);
  float nc    = fmaxf(n, EPSF);
  float scale = tanhf(nc) / nc * mk;
  #pragma unroll
  for (int i = 0; i < 4; ++i) acc[i] = fmaxf(acc[i] * scale, 0.f) * mk;
  *(f32x4*)(out + (size_t)node * DD + lane * 4) = acc;
}

// ---------------------------------------------------------------------------
extern "C" void kernel_launch(void* const* d_in, const int* in_sizes, int n_in,
                              void* d_out, int out_size, void* d_ws, size_t ws_size,
                              hipStream_t stream) {
  const float* node_repr = (const float*)d_in[0];
  const int*   adj       = (const int*)d_in[1];
  const float* weight    = (const float*)d_in[2];
  const float* mask      = (const float*)d_in[3];
  const float* W_pos     = (const float*)d_in[4];
  const float* W_neg     = (const float*)d_in[5];
  float*       out       = (float*)d_out;

  char* ws = (char*)d_ws;
  size_t o = 0;
  float*    xbuf = (float*)(ws + o);     o += (size_t)NN * DD * sizeof(float);     // 25.6 MB
  _Float16* msgP = (_Float16*)(ws + o);  o += (size_t)NN * DD * sizeof(_Float16);  // 12.8 MB
  _Float16* msgN = (_Float16*)(ws + o);  o += (size_t)NN * DD * sizeof(_Float16);  // 12.8 MB
  _Float16* WTp  = (_Float16*)(ws + o);  o += (size_t)DD * DD * sizeof(_Float16);  // 32 KB
  _Float16* WTn  = (_Float16*)(ws + o);  o += (size_t)DD * DD * sizeof(_Float16);  // 32 KB
  (void)o; (void)ws_size; (void)in_sizes; (void)n_in; (void)out_size;

  const int nodeBlocks = (NN + 7) / 8;        // 8 waves (nodes) per 256-thread block
  const int gemmBlocks = (NN / 16 + 7) / 8;   // 3125 M-tiles, 8 waves per block

  // ---- layer 0 ----
  prep_kernel  <<<nodeBlocks, 256, 0, stream>>>(node_repr, mask, xbuf, /*do_log=*/0);
  wcvt_kernel  <<<2, 256, 0, stream>>>(W_pos, W_neg, WTp, WTn, /*layer=*/0);
  gemm_kernel  <<<gemmBlocks, 256, 0, stream>>>(xbuf, WTp, WTn, msgP, msgN);
  gather_kernel<<<nodeBlocks, 256, 0, stream>>>(adj, weight, mask, msgP, msgN, xbuf);

  // ---- layer 1 ----
  prep_kernel  <<<nodeBlocks, 256, 0, stream>>>(xbuf, mask, xbuf, /*do_log=*/1);
  wcvt_kernel  <<<2, 256, 0, stream>>>(W_pos, W_neg, WTp, WTn, /*layer=*/1);
  gemm_kernel  <<<gemmBlocks, 256, 0, stream>>>(xbuf, WTp, WTn, msgP, msgN);
  gather_kernel<<<nodeBlocks, 256, 0, stream>>>(adj, weight, mask, msgP, msgN, out);
}
